// SEQuart_15126874816666
// MI455X (gfx1250) — matrix operand order repaired and checked by reading, not compiled, not run on hardware
//
#include <hip/hip_runtime.h>

typedef _Float16 v16h __attribute__((ext_vector_type(16)));
typedef _Float16 v8h  __attribute__((ext_vector_type(8)));
typedef float    v8f  __attribute__((ext_vector_type(8)));
typedef unsigned int u32x4 __attribute__((ext_vector_type(4)));

#define CCH   64
#define HW    16384
#define NBAT  16
#define DIMK  256

// ---------------------------------------------------------------------------
// Kernel 1: mean over H*W for each (tensor, b, c) plane.
// blockIdx.x = t*1024 + b*64 + c  (4 * 16 * 64 = 4096 blocks)
// ---------------------------------------------------------------------------
__global__ void __launch_bounds__(256) pool_mean_kernel(
    const float* __restrict__ low, const float* __restrict__ high,
    const float* __restrict__ flow, const float* __restrict__ fb,
    float* __restrict__ means)
{
    int bx  = blockIdx.x;
    int tid = threadIdx.x;
    int t   = bx >> 10;
    const float* src = (t == 0) ? low : (t == 1) ? high : (t == 2) ? flow : fb;
    const float4* p = (const float4*)(src + (size_t)(bx & 1023) * HW);
    float s = 0.f;
    #pragma unroll
    for (int i = 0; i < 16; ++i) {
        float4 v = p[i * 256 + tid];
        s += v.x + v.y + v.z + v.w;
    }
    __shared__ float red[256];
    red[tid] = s;
    __syncthreads();
    for (int off = 128; off > 0; off >>= 1) {
        if (tid < off) red[tid] += red[tid + off];
        __syncthreads();
    }
    if (tid == 0) means[bx] = red[0] * (1.0f / (float)HW);
}

// ---------------------------------------------------------------------------
// Kernel 2: tiny graph/SE math + build fused per-batch f16 weight matrix
// M_b[o][g*64+c] = Wg_g[o][c] + e_g[b][c] * (S[o][c] - Wg_g[o][c])
// One block per batch (16 blocks, 256 threads).
// ---------------------------------------------------------------------------
__global__ void __launch_bounds__(256) prep_kernel(
    const float* __restrict__ means,
    const float* __restrict__ Wgcn, const float* __restrict__ bgcn,
    const float* __restrict__ W1, const float* __restrict__ b1,
    const float* __restrict__ W2, const float* __restrict__ b2,
    const float* __restrict__ W3, const float* __restrict__ b3,
    const float* __restrict__ W4, const float* __restrict__ b4,
    const float* __restrict__ Wgate,
    _Float16* __restrict__ M)
{
    __shared__ float fc[256], tbuf[256], feat[256], evals[256];
    __shared__ float adj4[16], nrm[4];
    __shared__ float S[4096];
    int tid = threadIdx.x;
    int b   = blockIdx.x;

    fc[tid] = means[(tid >> 6) * 1024 + b * 64 + (tid & 63)];   // fc[i*64+c]
    __syncthreads();

    if (tid < 4) {                       // row norms
        float s = 0.f;
        for (int c = 0; c < 64; ++c) { float v = fc[tid * 64 + c]; s += v * v; }
        nrm[tid] = sqrtf(s);
    }
    __syncthreads();

    if (tid < 16) {                      // cosine adjacency 4x4
        int i = tid >> 2, j = tid & 3;
        float s = 0.f;
        for (int c = 0; c < 64; ++c) s += fc[i * 64 + c] * fc[j * 64 + c];
        adj4[tid] = s / (nrm[i] * nrm[j]);
    }
    __syncthreads();

    {                                    // adj @ fc
        int i = tid >> 6, c = tid & 63;
        float s = 0.f;
        #pragma unroll
        for (int j = 0; j < 4; ++j) s += adj4[i * 4 + j] * fc[j * 64 + c];
        tbuf[tid] = s;
    }
    __syncthreads();

    {                                    // relu((adj@fc) @ Wgcn + bgcn) -> feat_cat
        int i = tid >> 6, d = tid & 63;
        float s = bgcn[d];
        for (int c = 0; c < 64; ++c) s += tbuf[i * 64 + c] * Wgcn[c * 64 + d];
        feat[tid] = fmaxf(s, 0.f);
    }
    __syncthreads();

    {                                    // 4 sigmoid SE heads
        int k = tid >> 6, c = tid & 63;
        const float* Wk = (k == 0) ? W1 : (k == 1) ? W2 : (k == 2) ? W3 : W4;
        const float* bk = (k == 0) ? b1 : (k == 1) ? b2 : (k == 2) ? b3 : b4;
        float s = bk[c];
        for (int n = 0; n < 256; ++n) s += feat[n] * Wk[c * 256 + n];
        evals[tid] = 1.0f / (1.0f + __expf(-s));
    }
    // S[o*64+c] = sum over blocks of Wgate
    for (int it = 0; it < 16; ++it) {
        int idx = it * 256 + tid;
        int o = idx >> 6, c = idx & 63;
        S[idx] = Wgate[o * 256 + c] + Wgate[o * 256 + 64 + c]
               + Wgate[o * 256 + 128 + c] + Wgate[o * 256 + 192 + c];
    }
    __syncthreads();

    _Float16* Mb = M + (size_t)b * (64 * 256);
    for (int it = 0; it < 64; ++it) {
        int idx = it * 256 + tid;        // == o*256 + col
        int o = idx >> 8, col = idx & 255;
        int g = col >> 6, c = col & 63;
        float w = Wgate[idx];
        float coef = w + evals[g * 64 + c] * (S[o * 64 + c] - w);
        Mb[idx] = (_Float16)coef;
    }
}

// ---------------------------------------------------------------------------
// Kernel 3: batched GEMM  out_b[64 x 16384] = M_b[64 x 256] * X_b + bias
// using v_wmma_f32_16x16x32_f16. Grid (128 strips, 16 batches), 8 waves.
// Each wave owns one 16-column strip and all four 16-row tiles, so each X
// element is read from HBM exactly once.
// ---------------------------------------------------------------------------
__global__ void __launch_bounds__(256) gemm_kernel(
    const float* __restrict__ low, const float* __restrict__ high,
    const float* __restrict__ flow, const float* __restrict__ fb,
    const _Float16* __restrict__ M, const float* __restrict__ bgate,
    float* __restrict__ out)
{
    __shared__ _Float16 As[64 * 264];    // 64 rows, 256 halfs + 8 pad (33792 B)
    int tid   = threadIdx.x;
    int strip = blockIdx.x;              // 0..127
    int b     = blockIdx.y;              // 0..15

    // stage M_b into LDS with b128 copies
    const u32x4* gsrc = (const u32x4*)(M + (size_t)b * (64 * 256));
    #pragma unroll
    for (int it = 0; it < 8; ++it) {
        int flat = it * 256 + tid;       // 2048 x 16B
        int row  = flat >> 5;            // 32 u32x4 per row
        int q    = flat & 31;
        *((u32x4*)(&As[row * 264]) + q) = gsrc[flat];
    }
    __syncthreads();

    int wave = tid >> 5;
    int lane = tid & 31;
    int hsel = lane >> 4;                // half-wave select
    int nl   = lane & 15;
    int col  = strip * 128 + wave * 16 + nl;

    // accumulators pre-loaded with bias per C/D layout (VGPR i: M=i / i+8)
    v8f acc[4];
    #pragma unroll
    for (int rt = 0; rt < 4; ++rt)
        #pragma unroll
        for (int i = 0; i < 8; ++i)
            acc[rt][i] = bgate[rt * 16 + i + 8 * hsel];

    const float* xg[4] = { low, high, flow, fb };

    #pragma unroll
    for (int kk = 0; kk < 8; ++kk) {
        int k0 = kk * 32;
        int g  = kk >> 1;                        // source tensor (compile-time after unroll)
        int cbase = (k0 & 63) + 16 * hsel;       // first channel this half-wave reads
        const float* xcol = xg[g] + (size_t)(b * 64 + cbase) * HW + col;

        // B fragment: VGPR v holds K=2v,2v+1 (lanes 0-15) / K=2v+16,2v+17 (lanes 16-31)
        v16h bf;
        #pragma unroll
        for (int v = 0; v < 8; ++v) {
            float x0 = xcol[(2 * v) * HW];
            float x1 = xcol[(2 * v + 1) * HW];
            bf[2 * v]     = (_Float16)x0;
            bf[2 * v + 1] = (_Float16)x1;
        }
        if (kk < 7) {                            // prefetch next K-block (exact address)
            int g2 = (kk + 1) >> 1;
            int cb2 = ((k0 + 32) & 63) + 16 * hsel;
            __builtin_prefetch(xg[g2] + (size_t)(b * 64 + cb2) * HW + col, 0, 0);
        }

        // A fragments from LDS: lanes 0-15 K=k0..k0+15; lanes 16-31 K=k0+16..k0+31
        #pragma unroll
        for (int rt = 0; rt < 4; ++rt) {
            int arow = rt * 16 + nl;
            v8h alo = *(const v8h*)(&As[arow * 264 + k0 + 16 * hsel]);
            v8h ahi = *(const v8h*)(&As[arow * 264 + k0 + 8 + 16 * hsel]);
            v16h af = __builtin_shufflevector(alo, ahi,
                        0, 1, 2, 3, 4, 5, 6, 7, 8, 9, 10, 11, 12, 13, 14, 15);
            acc[rt] = __builtin_amdgcn_wmma_f32_16x16x32_f16(
                false, af, false, bf, (short)0, acc[rt], false, false);
        }
    }

    // write back per C/D layout
    #pragma unroll
    for (int rt = 0; rt < 4; ++rt)
        #pragma unroll
        for (int i = 0; i < 8; ++i) {
            int row = rt * 16 + i + 8 * hsel;
            out[(size_t)(b * 64 + row) * HW + col] = acc[rt][i];
        }
}

// ---------------------------------------------------------------------------
extern "C" void kernel_launch(void* const* d_in, const int* in_sizes, int n_in,
                              void* d_out, int out_size, void* d_ws, size_t ws_size,
                              hipStream_t stream)
{
    const float* low   = (const float*)d_in[0];
    const float* high  = (const float*)d_in[1];
    const float* flow  = (const float*)d_in[2];
    const float* fb    = (const float*)d_in[3];
    const float* Wgcn  = (const float*)d_in[4];
    const float* bgcn  = (const float*)d_in[5];
    const float* W1 = (const float*)d_in[6];   const float* b1 = (const float*)d_in[7];
    const float* W2 = (const float*)d_in[8];   const float* b2 = (const float*)d_in[9];
    const float* W3 = (const float*)d_in[10];  const float* b3 = (const float*)d_in[11];
    const float* W4 = (const float*)d_in[12];  const float* b4 = (const float*)d_in[13];
    const float* Wgate = (const float*)d_in[14];
    const float* bgate = (const float*)d_in[15];
    float* out = (float*)d_out;

    float*    means = (float*)d_ws;                         // 4*16*64 f32 = 16 KB
    _Float16* M     = (_Float16*)((char*)d_ws + 16384);     // 16*64*256 f16 = 512 KB

    pool_mean_kernel<<<4096, 256, 0, stream>>>(low, high, flow, fb, means);
    prep_kernel<<<16, 256, 0, stream>>>(means, Wgcn, bgcn,
                                        W1, b1, W2, b2, W3, b3, W4, b4, Wgate, M);
    gemm_kernel<<<dim3(128, 16), 256, 0, stream>>>(low, high, flow, fb, M, bgate, out);
}
